// Attention_63694364999844
// MI455X (gfx1250) — matrix operand order repaired and checked by reading, not compiled, not run on hardware
//
#include <hip/hip_runtime.h>

typedef __attribute__((ext_vector_type(16))) _Float16 v16h;
typedef __attribute__((ext_vector_type(8)))  float    v8f;
typedef __attribute__((ext_vector_type(4)))  float    v4f;

#define BDIM 64
#define TDIM 2048
#define HDIM 256
#define ADIM 4
#define BT   (BDIM * TDIM)
#define ROWS 32   /* rows of BT handled per workgroup in k1 */

// ---------------------------------------------------------------------------
// Kernel 0: pre-swizzle W1 (fp32 [K=256, N=256]) into f16 B-fragment order:
//   u32 index = ((s*16 + nt)*32 + lane)*8 + j
//   lane: n = nt*16 + (lane&15), hi = lane>>4 ; dword j holds K = s*32+2j+16hi (+1)
// 32768 dwords total (128 KiB), L2-resident for the whole GEMM.
// ---------------------------------------------------------------------------
__global__ __launch_bounds__(256)
void k0_swzW1(const float* __restrict__ W1, unsigned* __restrict__ w1swz)
{
    const int p  = blockIdx.x * 256 + threadIdx.x;    // 0 .. 32767
    const int j  = p & 7;
    const int L  = (p >> 3) & 31;
    const int nt = (p >> 8) & 15;
    const int s  = p >> 12;
    const int n  = nt * 16 + (L & 15);
    const int hi = L >> 4;
    const int k  = s * 32 + 2 * j + 16 * hi;
    union { _Float16 h[2]; unsigned u; } pk;
    pk.h[0] = (_Float16)W1[k * HDIM + n];
    pk.h[1] = (_Float16)W1[(k + 1) * HDIM + n];
    w1swz[p] = pk.u;
}

// ---------------------------------------------------------------------------
// Kernel 1: a[bt, 0..3] = tanh(X @ W1 + b1) @ W2 + b2
//   - X tile -> LDS in A-fragment order (f16)
//   - pre-swizzled W1 slabs -> LDS via double-buffered async-to-LDS copies
//   - GEMM1: 32 static v_wmma_f32_16x16x32_f16, fragments via ds_load_b128
// ---------------------------------------------------------------------------
__global__ __launch_bounds__(256)
void k1_proj(const float* __restrict__ X, const unsigned* __restrict__ w1swz,
             const float* __restrict__ b1, const float* __restrict__ W2,
             const float* __restrict__ b2, float* __restrict__ aOut)
{
    // A fragments: [s(8)][rg(2)][lane(32)][8 u32]  = 16 KiB
    __shared__ __align__(16) unsigned Xs[8 * 2 * 32 * 8];
    // B slabs, double buffered: [buf(2)][nt(16)][lane(32)][8 u32] = 32 KiB
    __shared__ __align__(16) unsigned Wbuf[2 * 16 * 32 * 8];

    const int tid  = threadIdx.x;
    const int lane = tid & 31;          // wave32
    const int wave = tid >> 5;          // 0..7
    const int rg   = wave >> 2;         // row group 0..1  (16 rows each)
    const int cg   = wave & 3;          // col group 0..3  (64 cols each)
    const int rlo  = lane & 15;
    const int hi   = lane >> 4;
    const long wgRow0 = (long)blockIdx.x * ROWS;

    const unsigned ldsWbuf = (unsigned)(unsigned long long)(void*)&Wbuf[0];

    // ---- issue async copy of slab 0 (4 x b128 per thread) -----------------
    #pragma unroll
    for (int i = 0; i < 4; ++i) {
        asm volatile("global_load_async_to_lds_b128 %0, %1, %2"
                     :: "v"(ldsWbuf + (unsigned)tid * 16u + (unsigned)i * 4096u),
                        "v"((unsigned)tid * 16u + (unsigned)i * 4096u),
                        "s"(w1swz)
                     : "memory");
    }

    // ---- stage X tile in A-fragment order (16 u32 per thread) -------------
    #pragma unroll
    for (int i = 0; i < 16; ++i) {
        const int g  = i * 256 + tid;        // ((s*2+rg)*32 + L)*8 + j
        const int j  = g & 7;
        const int L  = (g >> 3) & 31;
        const int rgg = (g >> 8) & 1;
        const int ss  = g >> 9;
        const int row = rgg * 16 + (L & 15);
        const int hh  = L >> 4;
        const int kloc = 2 * j + 16 * hh;
        const int k = ss * 32 + kloc;
        const float2 xv = *(const float2*)&X[(wgRow0 + row) * HDIM + k];
        union { _Float16 h[2]; unsigned u; } pk;
        pk.h[0] = (_Float16)xv.x;
        pk.h[1] = (_Float16)xv.y;
        Xs[g] = pk.u;
    }

    v8f acc[4] = {};   // each wave: 16 rows x 64 cols, fp32

    #pragma unroll
    for (int s = 0; s < 8; ++s) {
        asm volatile("s_wait_asynccnt 0" ::: "memory");
        __syncthreads();                          // slab s (and Xs) ready

        if (s < 7) {                              // prefetch slab s+1
            const unsigned dst = ldsWbuf + (unsigned)(((s + 1) & 1) * 16384);
            const unsigned src = (unsigned)((s + 1) * 16384);
            #pragma unroll
            for (int i = 0; i < 4; ++i) {
                asm volatile("global_load_async_to_lds_b128 %0, %1, %2"
                             :: "v"(dst + (unsigned)tid * 16u + (unsigned)i * 4096u),
                                "v"(src + (unsigned)tid * 16u + (unsigned)i * 4096u),
                                "s"(w1swz)
                             : "memory");
            }
        }

        // A fragment: 2 x ds_load_b128
        union { v16h v; uint4 q[2]; } A;
        const int abase = ((s * 2 + rg) * 32 + lane) * 8;
        A.q[0] = *(const uint4*)&Xs[abase];
        A.q[1] = *(const uint4*)&Xs[abase + 4];

        const unsigned sb = (s & 1) * 4096;       // u32 offset of current slab
        #pragma unroll
        for (int t = 0; t < 4; ++t) {             // 4 N-tiles per wave
            union { v16h v; uint4 q[2]; } B;
            const int bbase = sb + ((cg * 4 + t) * 32 + lane) * 8;
            B.q[0] = *(const uint4*)&Wbuf[bbase];
            B.q[1] = *(const uint4*)&Wbuf[bbase + 4];
            acc[t] = __builtin_amdgcn_wmma_f32_16x16x32_f16(
                         false, A.v, false, B.v, (short)0, acc[t], false, false);
        }
    }

    __syncthreads();                  // all waves done with Wbuf
    float* Hs = (float*)Wbuf;         // reuse LDS: hidden tile 32x256 fp32

    // Epilogue: + b1, tanh, scatter per C-layout (n = lane&15, m = v + 8*hi)
    #pragma unroll
    for (int t = 0; t < 4; ++t) {
        const int n    = cg * 64 + t * 16 + rlo;
        const float bv = b1[n];
        #pragma unroll
        for (int v = 0; v < 8; ++v) {
            const int m = rg * 16 + v + 8 * hi;
            Hs[m * HDIM + n] = tanhf(acc[t][v] + bv);
        }
    }
    __syncthreads();

    // GEMM2: a[row, kk] = hidden[row,:] . W2[:,kk] + b2[kk]  (tiny: 32x4)
    if (tid < ROWS * ADIM) {
        const int row = tid >> 2, kk = tid & 3;
        float s2 = b2[kk];
        for (int n = 0; n < HDIM; ++n)
            s2 += Hs[row * HDIM + n] * W2[n * ADIM + kk];
        aOut[(wgRow0 + row) * ADIM + kk] = s2;
    }
}

// ---------------------------------------------------------------------------
// Kernel 2: per (b, ch): amax over t, e = exp(a-amax), exclusive cumsum,
//           w = se_excl / (se_excl==0 ? 1 : se_excl)
// ---------------------------------------------------------------------------
__global__ __launch_bounds__(256)
void k2_scan(const float* __restrict__ aIn, float* __restrict__ wOut)
{
    __shared__ float sa[TDIM * ADIM];   // 32 KiB
    __shared__ float red[256];
    const int tid   = threadIdx.x;
    const long base = (long)blockIdx.x * TDIM * ADIM;

    for (int i = tid; i < TDIM * ADIM; i += 256) sa[i] = aIn[base + i];
    __syncthreads();

    for (int ch = 0; ch < ADIM; ++ch) {
        // max over time
        float m = -3.402823466e38f;
        #pragma unroll
        for (int j = 0; j < 8; ++j)
            m = fmaxf(m, sa[(tid * 8 + j) * ADIM + ch]);
        red[tid] = m;
        __syncthreads();
        for (int off = 128; off > 0; off >>= 1) {
            if (tid < off) red[tid] = fmaxf(red[tid], red[tid + off]);
            __syncthreads();
        }
        const float amax = red[0];
        __syncthreads();

        // stabilized exp + per-thread partials
        float ev[8], s = 0.f;
        #pragma unroll
        for (int j = 0; j < 8; ++j) {
            ev[j] = __expf(sa[(tid * 8 + j) * ADIM + ch] - amax);
            s += ev[j];
        }
        red[tid] = s;
        __syncthreads();
        if (tid == 0) {                     // exclusive scan of 256 block sums
            float off = 0.f;
            for (int i = 0; i < 256; ++i) { float tv = red[i]; red[i] = off; off += tv; }
        }
        __syncthreads();

        float run = red[tid];               // exclusive prefix at t = tid*8
        #pragma unroll
        for (int j = 0; j < 8; ++j) {
            const long t = tid * 8 + j;
            const float se = run;
            wOut[base + t * ADIM + ch] = se / ((se == 0.f) ? 1.f : se);
            run += ev[j];
        }
        __syncthreads();
    }
}

// ---------------------------------------------------------------------------
// Kernel 3: out[bt, h, kk] = X[bt, h] * w[bt, kk]
//   512 MiB streaming store -> non-temporal (ext-vector v4f for the builtin)
//   so W1/w stay L2-resident.
// ---------------------------------------------------------------------------
__global__ __launch_bounds__(256)
void k3_outer(const float* __restrict__ X, const float* __restrict__ w,
              float* __restrict__ out)
{
    const long idx = (long)blockIdx.x * 256 + threadIdx.x;   // over BT*H
    const long bt  = idx >> 8;
    const float x  = X[idx];
    const v4f wv = *(const v4f*)(w + bt * ADIM);
    v4f o;
    o.x = x * wv.x; o.y = x * wv.y; o.z = x * wv.z; o.w = x * wv.w;
    __builtin_nontemporal_store(o, (v4f*)out + idx);
}

// ---------------------------------------------------------------------------
extern "C" void kernel_launch(void* const* d_in, const int* in_sizes, int n_in,
                              void* d_out, int out_size, void* d_ws, size_t ws_size,
                              hipStream_t stream)
{
    (void)in_sizes; (void)n_in; (void)out_size; (void)ws_size;
    const float* X  = (const float*)d_in[0];
    const float* W1 = (const float*)d_in[1];
    const float* b1 = (const float*)d_in[2];
    const float* W2 = (const float*)d_in[3];
    const float* b2 = (const float*)d_in[4];

    float*    aBuf  = (float*)d_ws;                              // [BT,4] 2 MiB
    float*    wBuf  = aBuf + (size_t)BT * ADIM;                  // [BT,4] 2 MiB
    unsigned* w1swz = (unsigned*)((char*)d_ws + 4u * 1024u * 1024u); // 128 KiB

    k0_swzW1<<<128,                   256, 0, stream>>>(W1, w1swz);
    k1_proj <<<BT / ROWS,             256, 0, stream>>>(X, w1swz, b1, W2, b2, aBuf);
    k2_scan <<<BDIM,                  256, 0, stream>>>(aBuf, wBuf);
    k3_outer<<<(long)BT * HDIM / 256, 256, 0, stream>>>(X, wBuf, (float*)d_out);
}
